// CityExpertMoE_81561428951526
// MI455X (gfx1250) — matrix operand order repaired and checked by reading, not compiled, not run on hardware
//
#include <hip/hip_runtime.h>
#include <hip/hip_bf16.h>
#include <math.h>

// Problem constants (from reference)
#define BB 4
#define LL 2048
#define DD 1024
#define HH 4096
#define EE 8
#define NTOK (BB * LL)        // 8192 tokens
#define LN_EPS 1e-5f

// Tiling
#define TPT 32                // tokens per tile (2 x WMMA M)
#define BTHREADS 512          // 16 waves
#define HCHUNK 512            // H processed per chunk
#define XN_SW (DD + 8)        // padded LDS stride (bf16 elems) for xn tile
#define H_SW  (HCHUNK + 8)    // padded LDS stride for h chunk

typedef __bf16  v16bf __attribute__((ext_vector_type(16)));
typedef float   v8f   __attribute__((ext_vector_type(8)));

union FragB16 {
    v16bf v;
    uint4 q[2];   // 2 x 16B = 32B = 16 bf16
};

// ---------------------------------------------------------------------------
// Kernel 0: transpose + fp32->bf16 convert.  w:[E][R][C] -> wT:[E][C][R]
// (makes the WMMA B-operand K-dimension contiguous in memory)
// ---------------------------------------------------------------------------
__global__ void cvtT_kernel(const float* __restrict__ w, __bf16* __restrict__ wT,
                            int R, int C) {
    size_t total = (size_t)EE * R * C;
    for (size_t i = (size_t)blockIdx.x * blockDim.x + threadIdx.x; i < total;
         i += (size_t)gridDim.x * blockDim.x) {
        size_t e   = i / ((size_t)R * C);
        size_t rem = i - e * (size_t)R * C;
        size_t c   = rem / R;       // output: r fastest
        size_t r   = rem - c * R;
        wT[i] = (__bf16)w[(e * (size_t)R + r) * C + c];
    }
}

// ---------------------------------------------------------------------------
// Kernel 1: fused LayerNorm + router (softmax -> top2 -> renorm) per token.
// One 256-thread block per token.  Emits xn (bf16), residual into out,
// and per-expert gather lists.
// ---------------------------------------------------------------------------
__global__ __launch_bounds__(256) void ln_gate_kernel(
    const float* __restrict__ x, const float* __restrict__ gate_w,
    const float* __restrict__ gamma, const float* __restrict__ beta,
    __bf16* __restrict__ xnbf, int* __restrict__ counts,
    int* __restrict__ lists, float* __restrict__ wlists,
    float* __restrict__ out) {
    const int token = blockIdx.x;
    const int t = threadIdx.x;
    const float* xr = x + (size_t)token * DD;

    float v[4];
    float s = 0.f, s2 = 0.f;
#pragma unroll
    for (int i = 0; i < 4; ++i) {
        v[i] = xr[t * 4 + i];
        s += v[i];
        s2 += v[i] * v[i];
    }

    __shared__ float rs[256], rs2[256];
    __shared__ float gsum[EE * 256];
    rs[t] = s; rs2[t] = s2;
    __syncthreads();
#pragma unroll
    for (int o = 128; o > 0; o >>= 1) {
        if (t < o) { rs[t] += rs[t + o]; rs2[t] += rs2[t + o]; }
        __syncthreads();
    }
    const float mu = rs[0] * (1.f / DD);
    const float var = rs2[0] * (1.f / DD) - mu * mu;
    const float rstd = rsqrtf(var + LN_EPS);

    float pl[EE];
#pragma unroll
    for (int e = 0; e < EE; ++e) pl[e] = 0.f;
#pragma unroll
    for (int i = 0; i < 4; ++i) {
        const int d = t * 4 + i;
        const float xn = (v[i] - mu) * rstd * gamma[d] + beta[d];
        xnbf[(size_t)token * DD + d] = (__bf16)xn;
        out[(size_t)token * DD + d] = v[i];          // residual
#pragma unroll
        for (int e = 0; e < EE; ++e) pl[e] += xn * gate_w[d * EE + e];
    }
#pragma unroll
    for (int e = 0; e < EE; ++e) gsum[e * 256 + t] = pl[e];
    __syncthreads();
#pragma unroll
    for (int o = 128; o > 0; o >>= 1) {
        if (t < o) {
#pragma unroll
            for (int e = 0; e < EE; ++e) gsum[e * 256 + t] += gsum[e * 256 + t + o];
        }
        __syncthreads();
    }

    if (t == 0) {
        float l[EE];
#pragma unroll
        for (int e = 0; e < EE; ++e) l[e] = gsum[e * 256];
        // top-2
        int i0 = 0;
#pragma unroll
        for (int e = 1; e < EE; ++e) if (l[e] > l[i0]) i0 = e;
        int i1 = (i0 == 0) ? 1 : 0;
#pragma unroll
        for (int e = 0; e < EE; ++e)
            if (e != i0 && l[e] > l[i1]) i1 = e;
        // renormalized top-2 softmax == softmax over the two logits
        const float m = l[i0];
        const float p0 = expf(l[i0] - m), p1 = expf(l[i1] - m);
        const float inv = 1.f / (p0 + p1);
        const float w0 = p0 * inv, w1 = p1 * inv;
        int pos0 = atomicAdd(&counts[i0], 1);
        lists[i0 * NTOK + pos0] = token;
        wlists[i0 * NTOK + pos0] = w0;
        int pos1 = atomicAdd(&counts[i1], 1);
        lists[i1 * NTOK + pos1] = token;
        wlists[i1 * NTOK + pos1] = w1;
    }
}

// ---------------------------------------------------------------------------
// Kernel 2: per-(expert, 32-token tile) fused FFN:
//   h = gelu(Xe @ W1 + b1) ; out += cw * (h @ W2 + b2)
// bf16 WMMA, fp32 accumulate.  Async global->LDS for the gathered xn tile.
// 16 waves = 2 token-halves (M groups) x 8 column groups (N).
// ---------------------------------------------------------------------------
__device__ __forceinline__ v8f wmma_bf16(v16bf a, v16bf b, v8f c) {
    return __builtin_amdgcn_wmma_f32_16x16x32_bf16(false, a, false, b,
                                                   (short)0, c, false, false);
}

__global__ __launch_bounds__(BTHREADS, 1) void moe_ffn_kernel(
    const __bf16* __restrict__ xnbf, const __bf16* __restrict__ w1T,
    const __bf16* __restrict__ w2T, const float* __restrict__ b1,
    const float* __restrict__ b2, const int* __restrict__ counts,
    const int* __restrict__ lists, const float* __restrict__ wlists,
    float* __restrict__ out) {
    const int e = blockIdx.y;
    const int cnt = counts[e];
    const int row0 = blockIdx.x * TPT;
    if (row0 >= cnt) return;

    __shared__ __bf16 tileXn[TPT * XN_SW];   // 32 x 1032 bf16 = 66 KB
    __shared__ __bf16 hChunk[TPT * H_SW];    // 32 x 520  bf16 = 33 KB
    __shared__ int   sTok[TPT];
    __shared__ float sWgt[TPT];

    const int t = threadIdx.x;
    if (t < TPT) {
        const int r = row0 + t;
        const bool valid = r < cnt;
        sTok[t] = valid ? lists[e * NTOK + r] : 0;
        sWgt[t] = valid ? wlists[e * NTOK + r] : 0.f;
    }
    __syncthreads();

    // --- async gather of 32 xn rows (each thread: 128B of one row) ---
    {
        const int row = t >> 4;          // 0..31
        const int seg = t & 15;          // 0..15, 64 bf16 each
        const __bf16* src = xnbf + (size_t)sTok[row] * DD + seg * 64;
        unsigned lds0 = (unsigned)(uintptr_t)&tileXn[row * XN_SW + seg * 64];
#pragma unroll
        for (int i = 0; i < 8; ++i) {
            asm volatile("global_load_async_to_lds_b128 %0, %1, off"
                         :: "v"(lds0 + i * 16), "v"(src + i * 8)
                         : "memory");
        }
    }
    asm volatile("s_wait_asynccnt 0" ::: "memory");
    __syncthreads();

    const int wave = t >> 5;             // 0..15
    const int mg   = wave >> 3;          // token half: 0/1
    const int ng   = wave & 7;           // column group: 0..7
    const int lane = t & 31;
    const int cl = lane & 15;            // column within 16-wide tile
    const int kh = lane >> 4;            // K-half select
    const int rbase = mg * 16;           // this wave's 16-token half

    const __bf16* w1e = w1T + (size_t)e * HH * DD;   // [H][D]
    const __bf16* w2e = w2T + (size_t)e * DD * HH;   // [D][H]
    const float* b1e = b1 + e * HH;
    const float* b2e = b2 + e * DD;
    const int dbase = ng * 128;                      // this wave's 128 D-cols

    v8f acc[8];
#pragma unroll
    for (int i = 0; i < 8; ++i) acc[i] = {};

    for (int hc = 0; hc < HH; hc += HCHUNK) {
        // ---- GEMM1: wave computes 64 H columns for its token half ----
        v8f c1[4];
#pragma unroll
        for (int i = 0; i < 4; ++i) c1[i] = {};
        const int ncol0 = hc + ng * 64;
        for (int k = 0; k < DD; k += 32) {
            FragB16 a;
            const __bf16* ar = &tileXn[(rbase + cl) * XN_SW + k + kh * 16];
            a.q[0] = *(const uint4*)ar;
            a.q[1] = *(const uint4*)(ar + 8);
#pragma unroll
            for (int nt = 0; nt < 4; ++nt) {
                FragB16 b;
                const __bf16* bp =
                    w1e + (size_t)(ncol0 + nt * 16 + cl) * DD + k + kh * 16;
                b.q[0] = *(const uint4*)bp;
                b.q[1] = *(const uint4*)(bp + 8);
                c1[nt] = wmma_bf16(a.v, b.v, c1[nt]);
            }
        }
        // bias + exact-erf GELU, store bf16 h chunk to LDS
#pragma unroll
        for (int nt = 0; nt < 4; ++nt) {
            const float bv = b1e[ncol0 + nt * 16 + cl];
            const int col = ng * 64 + nt * 16 + cl;
#pragma unroll
            for (int vv = 0; vv < 8; ++vv) {
                const int row = rbase + vv + kh * 8;
                const float xg = c1[nt][vv] + bv;
                const float g =
                    0.5f * xg * (1.0f + erff(xg * 0.70710678118654752f));
                hChunk[row * H_SW + col] = (__bf16)g;
            }
        }
        __syncthreads();
        // ---- GEMM2 partial accumulate over this K chunk ----
        for (int k = 0; k < HCHUNK; k += 32) {
            FragB16 a;
            const __bf16* ar = &hChunk[(rbase + cl) * H_SW + k + kh * 16];
            a.q[0] = *(const uint4*)ar;
            a.q[1] = *(const uint4*)(ar + 8);
#pragma unroll
            for (int nt = 0; nt < 8; ++nt) {
                FragB16 b;
                const __bf16* bp =
                    w2e + (size_t)(dbase + nt * 16 + cl) * HH + hc + k + kh * 16;
                b.q[0] = *(const uint4*)bp;
                b.q[1] = *(const uint4*)(bp + 8);
                acc[nt] = wmma_bf16(a.v, b.v, acc[nt]);
            }
        }
        __syncthreads();
    }

    // ---- epilogue: weighted scatter-accumulate into output ----
#pragma unroll
    for (int nt = 0; nt < 8; ++nt) {
        const int dcol = dbase + nt * 16 + cl;
        const float bv = b2e[dcol];
#pragma unroll
        for (int vv = 0; vv < 8; ++vv) {
            const int row = rbase + vv + kh * 8;
            const float contrib = sWgt[row] * (acc[nt][vv] + bv);
            atomicAdd(&out[(size_t)sTok[row] * DD + dcol], contrib);
        }
    }
}

// ---------------------------------------------------------------------------
// Host-side launcher
// ---------------------------------------------------------------------------
extern "C" void kernel_launch(void* const* d_in, const int* in_sizes, int n_in,
                              void* d_out, int out_size, void* d_ws, size_t ws_size,
                              hipStream_t stream) {
    const float* x      = (const float*)d_in[0];   // [B,L,D]
    const float* gate_w = (const float*)d_in[1];   // [D,E]
    const float* w1     = (const float*)d_in[2];   // [E,D,H]
    const float* b1     = (const float*)d_in[3];   // [E,H]
    const float* w2     = (const float*)d_in[4];   // [E,H,D]
    const float* b2     = (const float*)d_in[5];   // [E,D]
    const float* gamma  = (const float*)d_in[6];   // [D]
    const float* beta   = (const float*)d_in[7];   // [D]
    float* out = (float*)d_out;                    // [B,L,D]

    // Workspace layout
    char* ws = (char*)d_ws;
    size_t off = 0;
    __bf16* w1T = (__bf16*)(ws + off); off += (size_t)EE * DD * HH * sizeof(__bf16);
    __bf16* w2T = (__bf16*)(ws + off); off += (size_t)EE * HH * DD * sizeof(__bf16);
    __bf16* xnbf = (__bf16*)(ws + off); off += (size_t)NTOK * DD * sizeof(__bf16);
    int* counts = (int*)(ws + off); off += 256;   // E ints, padded
    int* lists = (int*)(ws + off); off += (size_t)EE * NTOK * sizeof(int);
    float* wlists = (float*)(ws + off); off += (size_t)EE * NTOK * sizeof(float);

    hipMemsetAsync(counts, 0, EE * sizeof(int), stream);

    cvtT_kernel<<<4096, 256, 0, stream>>>(w1, w1T, DD, HH);  // -> [E][H][D]
    cvtT_kernel<<<4096, 256, 0, stream>>>(w2, w2T, HH, DD);  // -> [E][D][H]

    ln_gate_kernel<<<NTOK, 256, 0, stream>>>(x, gate_w, gamma, beta, xnbf,
                                             counts, lists, wlists, out);

    dim3 grid(NTOK / TPT, EE);   // tiles x experts; blocks past count early-exit
    moe_ffn_kernel<<<grid, BTHREADS, 0, stream>>>(xnbf, w1T, w2T, b1, b2,
                                                  counts, lists, wlists, out);
}
